// EnhancedAttentionLayer_89653147337499
// MI455X (gfx1250) — matrix operand order repaired and checked by reading, not compiled, not run on hardware
//
#include <hip/hip_runtime.h>
#include <hip/hip_bf16.h>
#include <math.h>

#define BATCH 2
#define SEQ   2048
#define DIMD  1536
#define HEADS 12
#define HD    128
#define MTOT  (BATCH*SEQ)   /* 4096 */

typedef __attribute__((ext_vector_type(16))) __bf16 v16bf;
typedef __attribute__((ext_vector_type(8)))  __bf16 v8bf;
typedef __attribute__((ext_vector_type(8)))  float  v8f;

__device__ __forceinline__ unsigned short f2bf_u(float f) {
  unsigned u = __float_as_uint(f);
  u += 0x7FFFu + ((u >> 16) & 1u);          // round-to-nearest-even
  return (unsigned short)(u >> 16);
}
__device__ __forceinline__ __bf16 f2bf(float f) {
  unsigned short s = f2bf_u(f);
  return __builtin_bit_cast(__bf16, s);
}

// ---------------------------------------------------------------- converts
__global__ void k_cvt_bf16(const float* __restrict__ in,
                           unsigned short* __restrict__ out, int n) {
  int i = blockIdx.x * blockDim.x + threadIdx.x;
  if (i < n) out[i] = f2bf_u(in[i]);
}

// ------------------------------------------------------------- rope tables
__global__ void k_rope_tables(float* __restrict__ cosT, float* __restrict__ sinT) {
  int i = blockIdx.x * blockDim.x + threadIdx.x;   // SEQ * (HD/2) threads
  int s = i >> 6, f = i & 63;
  float inv_freq = __expf(-((float)(2 * f) / (float)HD) * 9.210340371976184f); // ln(1e4)
  float ang = (float)s * inv_freq;
  cosT[i] = __cosf(ang);
  sinT[i] = __sinf(ang);
}

// ------------------------------------------------------------ bf16 WMMA GEMM
// Y[m][e] = sum_d A[m][d] * W[e][d] + bias[e]
// mode 0/1: + RoPE, write bf16 [B,H,S,hd]
// mode 2  : write bf16 transposed [B,H,hd,S]   (V)
// mode 3  : write f32 [M,N] (final output projection)
__global__ __launch_bounds__(256) void k_gemm_bf16_wmma(
    const unsigned short* __restrict__ Abf,
    const unsigned short* __restrict__ Wbf,
    const float* __restrict__ bias,
    unsigned short* __restrict__ outBf,
    float* __restrict__ outF,
    const float* __restrict__ cosT,
    const float* __restrict__ sinT,
    int mode)
{
  const int K = DIMD;
  int wid  = (int)((blockIdx.x * blockDim.x + threadIdx.x) >> 5);
  int lane = (int)(threadIdx.x & 31);
  int half = lane >> 4;
  int l16  = lane & 15;
  int mt   = wid & 255;          // M/16 = 256 tiles
  int nt64 = wid >> 8;           // N/64 = 24 supertiles
  int m0 = mt << 4;
  int n0 = nt64 << 6;

  v8f acc[4];
#pragma unroll
  for (int i = 0; i < 4; ++i)
#pragma unroll
    for (int j = 0; j < 8; ++j) acc[i][j] = 0.f;

  const unsigned short* arow = Abf + (size_t)(m0 + l16) * K;
  for (int kc = 0; kc < K; kc += 32) {
    // A fragment (16x32 bf16): lane<16 -> K 0..15 ; lane>=16 -> 16..31
    v16bf a;
    {
      v8bf lo = *(const v8bf*)(arow + kc + (half ? 16 : 0));
      v8bf hi = *(const v8bf*)(arow + kc + 8 + (half ? 16 : 0));
#pragma unroll
      for (int i = 0; i < 8; ++i) { a[i] = lo[i]; a[i + 8] = hi[i]; }
    }
#pragma unroll
    for (int nt = 0; nt < 4; ++nt) {
      // B fragment (32x16): lane = column e, 16 contiguous K values
      const unsigned short* bp =
          Wbf + (size_t)(n0 + nt * 16 + l16) * K + kc + (half ? 16 : 0);
      v16bf b = *(const v16bf*)bp;
      acc[nt] = __builtin_amdgcn_wmma_f32_16x16x32_bf16(
          false, a, false, b, (short)0, acc[nt], false, false);
    }
  }

  // epilogue
#pragma unroll
  for (int nt = 0; nt < 4; ++nt) {
    int e = n0 + nt * 16 + l16;
    float be = bias[e];
#pragma unroll
    for (int r = 0; r < 8; ++r) {
      int m = m0 + (half ? 8 + r : r);
      int b = m >> 11;            // / SEQ
      int s = m & (SEQ - 1);
      float v = acc[nt][r] + be;
      if (mode <= 1) {            // RoPE: pair (even,odd) = adjacent lanes
        float partner = __shfl_xor(v, 1, 32);
        int ein = e & (HD - 1);
        int fi  = ein >> 1;
        float c  = cosT[(s << 6) + fi];
        float sn = sinT[(s << 6) + fi];
        float ro = (e & 1) ? (partner * sn + v * c) : (v * c - partner * sn);
        int hh = e >> 7;
        size_t addr = (((size_t)b * HEADS + hh) * SEQ + s) * HD + ein;
        outBf[addr] = f2bf_u(ro);
      } else if (mode == 2) {     // V transposed [B,H,hd,S]
        int hh = e >> 7;
        int ein = e & (HD - 1);
        size_t addr = (((size_t)b * HEADS + hh) * HD + ein) * SEQ + s;
        outBf[addr] = f2bf_u(v);
      } else {                    // final projection, f32
        outF[(size_t)m * DIMD + e] = v;
      }
    }
  }
}

// ------------------------------------------------------- flash attention WMMA
// Computes S^T = K.Q^T so that exp(S^T) lands exactly in the A-fragment
// layout for the P x V WMMA (no cross-lane transpose needed).
__global__ __launch_bounds__(256) void k_attn_wmma(
    const unsigned short* __restrict__ Qh,   // [B,H,S,hd]
    const unsigned short* __restrict__ Kh,   // [B,H,S,hd]
    const unsigned short* __restrict__ Vt,   // [B,H,hd,S]
    unsigned short* __restrict__ ctx)        // [B,S,D] bf16
{
  int wid  = (int)((blockIdx.x * blockDim.x + threadIdx.x) >> 5);
  int lane = (int)(threadIdx.x & 31);
  int half = lane >> 4, l16 = lane & 15;
  int qt = wid & 127;            // S/16 = 128 q-tiles
  int bh = wid >> 7;             // 0..23
  int b = bh / HEADS, h = bh % HEADS;
  const float scale = 0.088388347648318447f;     // 1/sqrt(128)
  // ALiBi slopes for H=12: h<8 -> 2^-(h+1) ; h>=8 -> 2^-((h-8)+0.5)
  float slope = (h < 8) ? exp2f(-(float)(h + 1)) : exp2f(-((float)(h - 8) + 0.5f));

  const unsigned short* Qb = Qh + (size_t)bh * SEQ * HD;
  const unsigned short* Kb = Kh + (size_t)bh * SEQ * HD;
  const unsigned short* Vb = Vt + (size_t)bh * HD * SEQ;
  int q0 = qt << 4;

  // Q^T B-fragments, one per 32-wide d chunk (lane = column q)
  v16bf bq[4];
#pragma unroll
  for (int dc = 0; dc < 4; ++dc)
    bq[dc] = *(const v16bf*)(Qb + (size_t)(q0 + l16) * HD + dc * 32 + half * 16);

  v8f out[8];
#pragma unroll
  for (int i = 0; i < 8; ++i)
#pragma unroll
    for (int j = 0; j < 8; ++j) out[i][j] = 0.f;
  float m_run = -1e30f, l_run = 0.f;

  for (int kc = 0; kc < SEQ; kc += 32) {
    v8f c0, c1;
#pragma unroll
    for (int j = 0; j < 8; ++j) { c0[j] = 0.f; c1[j] = 0.f; }
#pragma unroll
    for (int dc = 0; dc < 4; ++dc) {
      v16bf a0, a1;
      const unsigned short* k0 = Kb + (size_t)(kc + l16) * HD + dc * 32;
      const unsigned short* k1 = Kb + (size_t)(kc + 16 + l16) * HD + dc * 32;
      {
        v8bf lo = *(const v8bf*)(k0 + (half ? 16 : 0));
        v8bf hi = *(const v8bf*)(k0 + 8 + (half ? 16 : 0));
#pragma unroll
        for (int i = 0; i < 8; ++i) { a0[i] = lo[i]; a0[i + 8] = hi[i]; }
      }
      {
        v8bf lo = *(const v8bf*)(k1 + (half ? 16 : 0));
        v8bf hi = *(const v8bf*)(k1 + 8 + (half ? 16 : 0));
#pragma unroll
        for (int i = 0; i < 8; ++i) { a1[i] = lo[i]; a1[i + 8] = hi[i]; }
      }
      c0 = __builtin_amdgcn_wmma_f32_16x16x32_bf16(false, a0, false, bq[dc],
                                                   (short)0, c0, false, false);
      c1 = __builtin_amdgcn_wmma_f32_16x16x32_bf16(false, a1, false, bq[dc],
                                                   (short)0, c1, false, false);
    }
    // c0/c1 hold S^T: row k = kc + (half?8:0) + r (+16 for c1), col q = l16
    float s0[8], s1[8], cmax = -1e30f;
    int kb0 = kc + (half ? 8 : 0);
#pragma unroll
    for (int r = 0; r < 8; ++r) {
      s0[r] = c0[r] * scale + slope * (float)(kb0 + r);
      s1[r] = c1[r] * scale + slope * (float)(kb0 + 16 + r);
      cmax = fmaxf(cmax, fmaxf(s0[r], s1[r]));
    }
    cmax = fmaxf(cmax, __shfl_xor(cmax, 16, 32));     // join k halves per q
    float m_new = fmaxf(m_run, cmax);
    float alpha = __expf(m_run - m_new);
    v16bf pa;                     // P as 16x32 A fragment (q rows, k contraction)
    float psum = 0.f;
#pragma unroll
    for (int r = 0; r < 8; ++r) {
      float p0 = __expf(s0[r] - m_new);
      float p1 = __expf(s1[r] - m_new);
      psum += p0 + p1;
      pa[r]     = f2bf(p0);
      pa[r + 8] = f2bf(p1);
    }
    psum += __shfl_xor(psum, 16, 32);
    l_run = l_run * alpha + psum;
    m_run = m_new;
#pragma unroll
    for (int nt = 0; nt < 8; ++nt) {
#pragma unroll
      for (int r = 0; r < 8; ++r) out[nt][r] *= alpha;
      // V^T fragment: lane = output column n, k = 8h..8h+7 and 16+8h..16+8h+7
      v16bf bv;
      {
        const unsigned short* vp = Vb + (size_t)(nt * 16 + l16) * SEQ + kc + half * 8;
        v8bf lo = *(const v8bf*)vp;
        v8bf hi = *(const v8bf*)(vp + 16);
#pragma unroll
        for (int i = 0; i < 8; ++i) { bv[i] = lo[i]; bv[i + 8] = hi[i]; }
      }
      out[nt] = __builtin_amdgcn_wmma_f32_16x16x32_bf16(
          false, pa, false, bv, (short)0, out[nt], false, false);
    }
  }

  // normalize: l_run lives per-lane at q = l16 (replicated in both halves);
  // output rows are q = q0 + r (lane<16) / q0+8+r (lane>=16)
  float rl[8];
#pragma unroll
  for (int r = 0; r < 8; ++r) {
    float lo = __shfl(l_run, r, 32);
    float hi = __shfl(l_run, r + 8, 32);
    rl[r] = 1.f / (half ? hi : lo);
  }
#pragma unroll
  for (int nt = 0; nt < 8; ++nt) {
    int e = h * HD + nt * 16 + l16;
#pragma unroll
    for (int r = 0; r < 8; ++r) {
      int q = q0 + (half ? 8 + r : r);
      ctx[((size_t)b * SEQ + q) * DIMD + e] = f2bf_u(out[nt][r] * rl[r]);
    }
  }
}

// -------------------------------------------------------------------- launch
extern "C" void kernel_launch(void* const* d_in, const int* in_sizes, int n_in,
                              void* d_out, int out_size, void* d_ws, size_t ws_size,
                              hipStream_t stream) {
  (void)in_sizes; (void)n_in; (void)out_size; (void)ws_size;
  const float* x  = (const float*)d_in[0];
  const float* Wq = (const float*)d_in[1];
  const float* bq = (const float*)d_in[2];
  const float* Wk = (const float*)d_in[3];
  const float* bk = (const float*)d_in[4];
  const float* Wv = (const float*)d_in[5];
  const float* bv = (const float*)d_in[6];
  const float* Wo = (const float*)d_in[7];
  const float* bo = (const float*)d_in[8];
  float* out = (float*)d_out;

  char* p = (char*)d_ws;
  auto take = [&](size_t bytes) -> char* {
    char* r = p;
    p += (bytes + 255) & ~(size_t)255;
    return r;
  };
  unsigned short* xb   = (unsigned short*)take((size_t)MTOT * DIMD * 2);
  unsigned short* wqb  = (unsigned short*)take((size_t)DIMD * DIMD * 2);
  unsigned short* wkb  = (unsigned short*)take((size_t)DIMD * DIMD * 2);
  unsigned short* wvb  = (unsigned short*)take((size_t)DIMD * DIMD * 2);
  unsigned short* wob  = (unsigned short*)take((size_t)DIMD * DIMD * 2);
  unsigned short* qh   = (unsigned short*)take((size_t)MTOT * DIMD * 2);  // [B,H,S,hd]
  unsigned short* kh   = (unsigned short*)take((size_t)MTOT * DIMD * 2);  // [B,H,S,hd]
  unsigned short* vT   = (unsigned short*)take((size_t)MTOT * DIMD * 2);  // [B,H,hd,S]
  unsigned short* ctxb = (unsigned short*)take((size_t)MTOT * DIMD * 2);  // [B,S,D]
  float* cosT = (float*)take((size_t)SEQ * 64 * 4);
  float* sinT = (float*)take((size_t)SEQ * 64 * 4);

  const int nX = MTOT * DIMD;       // 6291456
  const int nW = DIMD * DIMD;       // 2359296
  k_cvt_bf16<<<(nX + 255) / 256, 256, 0, stream>>>(x,  xb,  nX);
  k_cvt_bf16<<<(nW + 255) / 256, 256, 0, stream>>>(Wq, wqb, nW);
  k_cvt_bf16<<<(nW + 255) / 256, 256, 0, stream>>>(Wk, wkb, nW);
  k_cvt_bf16<<<(nW + 255) / 256, 256, 0, stream>>>(Wv, wvb, nW);
  k_cvt_bf16<<<(nW + 255) / 256, 256, 0, stream>>>(Wo, wob, nW);
  k_rope_tables<<<(SEQ * 64) / 256, 256, 0, stream>>>(cosT, sinT);

  // 6144 wave-tiles / 8 waves per block = 768 blocks
  k_gemm_bf16_wmma<<<768, 256, 0, stream>>>(xb, wqb, bq, qh, nullptr, cosT, sinT, 0);
  k_gemm_bf16_wmma<<<768, 256, 0, stream>>>(xb, wkb, bk, kh, nullptr, cosT, sinT, 1);
  k_gemm_bf16_wmma<<<768, 256, 0, stream>>>(xb, wvb, bv, vT, nullptr, cosT, sinT, 2);

  // 3072 wave-tiles / 8 = 384 blocks
  k_attn_wmma<<<384, 256, 0, stream>>>(qh, kh, vT, ctxb);

  k_gemm_bf16_wmma<<<768, 256, 0, stream>>>(ctxb, wob, bo, nullptr, out, cosT, sinT, 3);
}
